// RF_55035710931677
// MI455X (gfx1250) — matrix operand order repaired and checked by reading, not compiled, not run on hardware
//
#include <hip/hip_runtime.h>

#define L_LEN 16384
#define P_DIM 512
#define NCHUNK 128
#define CHLEN 128   // NCHUNK * CHLEN == L_LEN

typedef __attribute__((ext_vector_type(16))) __bf16 v16bf;
typedef __attribute__((ext_vector_type(8)))  __bf16 v8bf;
typedef __attribute__((ext_vector_type(8)))  float  v8f;

// ---------- helpers ----------
__device__ __forceinline__ unsigned short f2bf(float f) {
    unsigned int u = __float_as_uint(f);
    unsigned int r = u + 0x7FFFu + ((u >> 16) & 1u);   // round-to-nearest-even
    return (unsigned short)(r >> 16);
}

// ---------- pass 0: per-channel ZOH constants ----------
// a = exp((lre + i*lim)*dt), b = (a-1)/(lre+i*lim), aL = a^CHLEN
__global__ __launch_bounds__(P_DIM) void rf_consts(
    const float* __restrict__ Lambda, const float* __restrict__ log_step,
    float* __restrict__ cAr, float* __restrict__ cAi,
    float* __restrict__ cBr, float* __restrict__ cBi,
    float* __restrict__ cLr, float* __restrict__ cLi)
{
    int p = threadIdx.x;
    float lre = -__expf(Lambda[2 * p]);
    float lim = Lambda[2 * p + 1];
    float dt  = __expf(log_step[p]);
    float mag = __expf(lre * dt);
    float ar  = mag * __cosf(lim * dt);
    float ai  = mag * __sinf(lim * dt);
    float den = lre * lre + lim * lim;
    float nr  = ar - 1.0f, ni = ai;
    float br  = (nr * lre + ni * lim) / den;
    float bi  = (ni * lre - nr * lim) / den;
    // a^128 via 7 complex squarings
    float pr = ar, pi = ai;
#pragma unroll
    for (int s = 0; s < 7; ++s) {
        float qr = pr * pr - pi * pi;
        float qi = 2.0f * pr * pi;
        pr = qr; pi = qi;
    }
    cAr[p] = ar; cAi[p] = ai;
    cBr[p] = br; cBi[p] = bi;
    cLr[p] = pr; cLi[p] = pi;
}

// ---------- pass 1: per-chunk final state with zero init ----------
__global__ __launch_bounds__(P_DIM) void rf_chunk_final(
    const float* __restrict__ u,
    const float* __restrict__ cAr, const float* __restrict__ cAi,
    const float* __restrict__ cBr, const float* __restrict__ cBi,
    float* __restrict__ fRe, float* __restrict__ fIm)
{
    int p = threadIdx.x;
    int c = blockIdx.x;
    float ar = cAr[p], ai = cAi[p], br = cBr[p], bi = cBi[p];
    float xr = 0.0f, xi = 0.0f;
    const float* up = u + (size_t)c * CHLEN * P_DIM + p;
#pragma unroll 4
    for (int t = 0; t < CHLEN; ++t) {
        float uv = up[(size_t)t * P_DIM];
        float nr = fmaf(ar, xr, fmaf(-ai, xi, br * uv));
        float ni = fmaf(ar, xi, fmaf( ai, xr, bi * uv));
        xr = nr; xi = ni;
    }
    fRe[c * P_DIM + p] = xr;
    fIm[c * P_DIM + p] = xi;
}

// ---------- pass 2: carry scan across chunks (sequential, tiny) ----------
__global__ __launch_bounds__(P_DIM) void rf_carry(
    const float* __restrict__ fRe, const float* __restrict__ fIm,
    const float* __restrict__ cLr, const float* __restrict__ cLi,
    float* __restrict__ gRe, float* __restrict__ gIm)
{
    int p = threadIdx.x;
    float lr = cLr[p], li = cLi[p];
    float cr = 0.0f, ci = 0.0f;
    for (int c = 0; c < NCHUNK; ++c) {
        gRe[c * P_DIM + p] = cr;
        gIm[c * P_DIM + p] = ci;
        float nr = fmaf(lr, cr, fmaf(-li, ci, fRe[c * P_DIM + p]));
        float ni = fmaf(lr, ci, fmaf( li, cr, fIm[c * P_DIM + p]));
        cr = nr; ci = ni;
    }
}

// ---------- pass 3: replay chunks seeded with carry, emit bf16 states ----------
__global__ __launch_bounds__(P_DIM) void rf_scan_write(
    const float* __restrict__ u,
    const float* __restrict__ cAr, const float* __restrict__ cAi,
    const float* __restrict__ cBr, const float* __restrict__ cBi,
    const float* __restrict__ gRe, const float* __restrict__ gIm,
    unsigned short* __restrict__ xRe, unsigned short* __restrict__ xIm)
{
    int p = threadIdx.x;
    int c = blockIdx.x;
    float ar = cAr[p], ai = cAi[p], br = cBr[p], bi = cBi[p];
    float xr = gRe[c * P_DIM + p];
    float xi = gIm[c * P_DIM + p];
    size_t base = (size_t)c * CHLEN * P_DIM + p;
#pragma unroll 4
    for (int t = 0; t < CHLEN; ++t) {
        float uv = u[base + (size_t)t * P_DIM];
        float nr = fmaf(ar, xr, fmaf(-ai, xi, br * uv));
        float ni = fmaf(ar, xi, fmaf( ai, xr, bi * uv));
        xr = nr; xi = ni;
        xRe[base + (size_t)t * P_DIM] = f2bf(xr);
        xIm[base + (size_t)t * P_DIM] = f2bf(xi);
    }
}

// ---------- pass 4: convert V to bf16 (V_im pre-negated) ----------
__global__ __launch_bounds__(256) void rf_conv_v(
    const float* __restrict__ Vre, const float* __restrict__ Vim,
    unsigned short* __restrict__ wRe, unsigned short* __restrict__ wImN)
{
    int i = blockIdx.x * 256 + threadIdx.x;
    if (i < P_DIM * P_DIM) {
        wRe[i]  = f2bf(Vre[i]);
        wImN[i] = f2bf(-Vim[i]);
    }
}

// ---------- pass 5: WMMA GEMM  Re(xs @ V^T)  + spike threshold ----------
// A-tile (16x32 bf16): lane = half*16 + m; element e -> p = k + half*16 + e.
//                      Two contiguous b128 loads.
__device__ __forceinline__ v16bf load_a(const unsigned short* __restrict__ X,
                                        int row, int k, int half)
{
    const __bf16* bp = (const __bf16*)(const void*)(X + (size_t)row * P_DIM);
    v8bf lo = *(const v8bf*)(const void*)(bp + k + half * 16);
    v8bf hi = *(const v8bf*)(const void*)(bp + k + half * 16 + 8);
    v16bf a;
#pragma unroll
    for (int i = 0; i < 8; ++i) { a[i] = lo[i]; a[i + 8] = hi[i]; }
    return a;
}

// B-tile (32x16 bf16, B[k][n] = W[jbase+n][kbase+k]): lane = half*16 + n;
// element e -> K = half*16 + e  -> 16 contiguous bf16 of W row (jbase+n).
__device__ __forceinline__ v16bf load_b(const unsigned short* __restrict__ W,
                                        int j, int k, int half)
{
    return *(const v16bf*)(const void*)(W + (size_t)j * P_DIM + k + half * 16);
}

// Each wave computes a 64(M) x 64(N) output tile: 4x4 grid of 16x16 WMMA tiles,
// 16 v8f accumulators. 8 waves per block arranged 2(M-groups) x 4(N-groups):
// block covers 128(M) x 256(N). Grid = (L/128, P/256).
// Per K-step per wave: 16 A b128 loads + 16 B b128 loads feed 32 WMMAs (1:1).
__global__ __launch_bounds__(256) void rf_gemm_spike(
    const unsigned short* __restrict__ xRe, const unsigned short* __restrict__ xIm,
    const unsigned short* __restrict__ wRe, const unsigned short* __restrict__ wImN,
    float* __restrict__ out)
{
    const int lane  = threadIdx.x & 31;
    const int wave  = threadIdx.x >> 5;        // 0..7
    const int waveM = wave >> 2;               // 0..1
    const int waveN = wave & 3;                // 0..3
    const int half  = lane >> 4;
    const int sub   = lane & 15;               // A row sub-index / B,C column
    const int lbase = blockIdx.x * 128 + waveM * 64;
    const int jbase = blockIdx.y * 256 + waveN * 64;

    v8f acc[4][4];
#pragma unroll
    for (int mi = 0; mi < 4; ++mi)
#pragma unroll
        for (int ni = 0; ni < 4; ++ni)
            acc[mi][ni] = (v8f){};

    for (int k = 0; k < P_DIM; k += 32) {
        v16bf aR[4], aI[4];
#pragma unroll
        for (int mi = 0; mi < 4; ++mi) {
            aR[mi] = load_a(xRe, lbase + mi * 16 + sub, k, half);
            aI[mi] = load_a(xIm, lbase + mi * 16 + sub, k, half);
        }
#pragma unroll
        for (int ni = 0; ni < 4; ++ni) {
            v16bf bR = load_b(wRe,  jbase + ni * 16 + sub, k, half);
            v16bf bI = load_b(wImN, jbase + ni * 16 + sub, k, half);
#pragma unroll
            for (int mi = 0; mi < 4; ++mi) {
                acc[mi][ni] = __builtin_amdgcn_wmma_f32_16x16x32_bf16(
                    false, aR[mi], false, bR, (short)0, acc[mi][ni], false, false);
                acc[mi][ni] = __builtin_amdgcn_wmma_f32_16x16x32_bf16(
                    false, aI[mi], false, bI, (short)0, acc[mi][ni], false, false);
            }
        }
    }

    // C/D layout: lane = half*16 + n ; VGPR r holds row m = r + 8*half
#pragma unroll
    for (int mi = 0; mi < 4; ++mi) {
#pragma unroll
        for (int r = 0; r < 8; ++r) {
            int m = r + 8 * half;
            size_t o = (size_t)(lbase + mi * 16 + m) * P_DIM + jbase + sub;
#pragma unroll
            for (int ni = 0; ni < 4; ++ni)
                out[o + ni * 16] = (acc[mi][ni][r] > 1.0f) ? 1.0f : 0.0f;
        }
    }
}

// ---------- launcher ----------
extern "C" void kernel_launch(void* const* d_in, const int* in_sizes, int n_in,
                              void* d_out, int out_size, void* d_ws, size_t ws_size,
                              hipStream_t stream)
{
    (void)in_sizes; (void)n_in; (void)out_size; (void)ws_size;
    const float* u        = (const float*)d_in[0];   // (L, P)
    const float* Lambda   = (const float*)d_in[1];   // (P, 2)
    const float* log_step = (const float*)d_in[2];   // (P, 1)
    const float* V_re     = (const float*)d_in[3];   // (P, P)
    const float* V_im     = (const float*)d_in[4];   // (P, P)
    float* out            = (float*)d_out;           // (L, P) float32 spikes

    char* ws = (char*)d_ws;
    auto alloc = [&](size_t bytes) -> void* {
        void* p = (void*)ws;
        ws += (bytes + 255) & ~(size_t)255;
        return p;
    };

    float* cAr = (float*)alloc(P_DIM * sizeof(float));
    float* cAi = (float*)alloc(P_DIM * sizeof(float));
    float* cBr = (float*)alloc(P_DIM * sizeof(float));
    float* cBi = (float*)alloc(P_DIM * sizeof(float));
    float* cLr = (float*)alloc(P_DIM * sizeof(float));
    float* cLi = (float*)alloc(P_DIM * sizeof(float));
    float* fRe = (float*)alloc((size_t)NCHUNK * P_DIM * sizeof(float));
    float* fIm = (float*)alloc((size_t)NCHUNK * P_DIM * sizeof(float));
    float* gRe = (float*)alloc((size_t)NCHUNK * P_DIM * sizeof(float));
    float* gIm = (float*)alloc((size_t)NCHUNK * P_DIM * sizeof(float));
    unsigned short* xRe  = (unsigned short*)alloc((size_t)L_LEN * P_DIM * sizeof(unsigned short));
    unsigned short* xIm  = (unsigned short*)alloc((size_t)L_LEN * P_DIM * sizeof(unsigned short));
    unsigned short* wRe  = (unsigned short*)alloc((size_t)P_DIM * P_DIM * sizeof(unsigned short));
    unsigned short* wImN = (unsigned short*)alloc((size_t)P_DIM * P_DIM * sizeof(unsigned short));

    rf_consts<<<1, P_DIM, 0, stream>>>(Lambda, log_step, cAr, cAi, cBr, cBi, cLr, cLi);
    rf_conv_v<<<(P_DIM * P_DIM + 255) / 256, 256, 0, stream>>>(V_re, V_im, wRe, wImN);
    rf_chunk_final<<<NCHUNK, P_DIM, 0, stream>>>(u, cAr, cAi, cBr, cBi, fRe, fIm);
    rf_carry<<<1, P_DIM, 0, stream>>>(fRe, fIm, cLr, cLi, gRe, gIm);
    rf_scan_write<<<NCHUNK, P_DIM, 0, stream>>>(u, cAr, cAi, cBr, cBi, gRe, gIm, xRe, xIm);

    dim3 grid(L_LEN / 128, P_DIM / 256);
    rf_gemm_spike<<<grid, 256, 0, stream>>>(xRe, xIm, wRe, wImN, out);
}
